// CTIBlock1_71536975282228
// MI455X (gfx1250) — matrix operand order repaired and checked by reading, not compiled, not run on hardware
//
#include <hip/hip_runtime.h>

typedef __attribute__((ext_vector_type(16))) __bf16 v16bf;
typedef __attribute__((ext_vector_type(8)))  float  v8f;

// ---------- bf16 helpers ----------
__device__ __forceinline__ unsigned short f32_to_bf16(float f) {
  union { float f; unsigned int u; } t; t.f = f;
  unsigned int u = t.u;
  unsigned int r = u + 0x7FFFu + ((u >> 16) & 1u); // round-to-nearest-even
  return (unsigned short)(r >> 16);
}
__device__ __forceinline__ float bf16_to_f32(unsigned short h) {
  union { unsigned int u; float f; } t; t.u = ((unsigned int)h) << 16;
  return t.f;
}

// ---------- elementwise f32 -> bf16 ----------
__global__ void cvt_bf16_kernel(const float* __restrict__ in,
                                unsigned short* __restrict__ out, size_t n) {
  size_t i = (size_t)blockIdx.x * blockDim.x + threadIdx.x;
  size_t stride = (size_t)gridDim.x * blockDim.x;
  for (; i < n; i += stride) out[i] = f32_to_bf16(in[i]);
}

// ---------- build padded [W_off | W_attn | 0] (384 x 224) + bias ----------
__global__ void build_wcat_kernel(const float* __restrict__ Woff, const float* __restrict__ boff,
                                  const float* __restrict__ Wattn, const float* __restrict__ battn,
                                  float* __restrict__ Wcat, float* __restrict__ bcat) {
  int tid = blockIdx.x * blockDim.x + threadIdx.x;
  const int total = 384 * 224;
  if (tid < total) {
    int k = tid / 224, n = tid % 224;
    float v = 0.f;
    if (n < 144) v = Woff[k * 144 + n];
    else if (n < 216) v = Wattn[k * 72 + (n - 144)];
    Wcat[tid] = v;
  }
  if (tid < 224) {
    float v = 0.f;
    if (tid < 144) v = boff[tid];
    else if (tid < 216) v = battn[tid - 144];
    bcat[tid] = v;
  }
}

// ---------- pack W (K x Npad row-major f32) into WMMA bf16 B-fragment layout ----------
// Layout: Wp[((kt*NT + nt)*32 + lane)*8 + v] = {bf16(W[k0][n]), bf16(W[k0+1][n])}
// with n = nt*16 + (lane&15), k0 = kt*32 + 2*v + 16*(lane>>4)   (32x16 B tile)
__global__ void pack_w_kernel(const float* __restrict__ W, unsigned int* __restrict__ Wp,
                              int K, int Npad) {
  int KT = K >> 5, NT = Npad >> 4;
  int total = KT * NT * 32 * 8;
  int tid = blockIdx.x * blockDim.x + threadIdx.x;
  if (tid >= total) return;
  int v    = tid & 7;
  int lane = (tid >> 3) & 31;
  int nt   = (tid >> 8) % NT;
  int kt   = (tid >> 8) / NT;
  int hi = lane >> 4, mr = lane & 15;
  int n  = nt * 16 + mr;
  int k0 = kt * 32 + 2 * v + 16 * hi;
  unsigned short lo = f32_to_bf16(W[(size_t)k0 * Npad + n]);
  unsigned short hh = f32_to_bf16(W[(size_t)(k0 + 1) * Npad + n]);
  Wp[tid] = (unsigned int)lo | ((unsigned int)hh << 16);
}

// ---------- reference-point processing (B blocks) ----------
__global__ void ref_kernel(const float* __restrict__ rp, float* __restrict__ ref) {
  int b = blockIdx.x;
  int tid = threadIdx.x;
  __shared__ float sx[256];
  __shared__ float sy[256];
  float ax = 0.f, ay = 0.f;
  for (int i = tid; i < 2304; i += 256) {
    ax += rp[((size_t)b * 3024 + i) * 2 + 0];
    ay += rp[((size_t)b * 3024 + i) * 2 + 1];
  }
  sx[tid] = ax; sy[tid] = ay;
  __syncthreads();
  for (int s = 128; s > 0; s >>= 1) {
    if (tid < s) { sx[tid] += sx[tid + s]; sy[tid] += sy[tid + s]; }
    __syncthreads();
  }
  float p1x = sx[0] * (1.f / 2304.f);
  float p1y = sy[0] * (1.f / 2304.f);
  for (int q = tid; q < 576; q += 256) {
    float p2x = rp[((size_t)b * 3024 + 2304 + q) * 2 + 0];
    float p2y = rp[((size_t)b * 3024 + 2304 + q) * 2 + 1];
    int i = q / 24, j = q % 24;
    float syc = (i + 0.5f) * 0.5f - 0.5f;  // 24 -> 12 half-pixel mapping
    float sxc = (j + 0.5f) * 0.5f - 0.5f;
    float fy = floorf(syc), fx = floorf(sxc);
    float ly = syc - fy, lx = sxc - fx;
    int y0 = (int)fy, x0 = (int)fx;
    int y0c = min(max(y0, 0), 11), y1c = min(max(y0 + 1, 0), 11);
    int x0c = min(max(x0, 0), 11), x1c = min(max(x0 + 1, 0), 11);
    const float* r3 = rp + ((size_t)b * 3024 + 2880) * 2;
    float p3x = (1.f - lx) * (1.f - ly) * r3[(y0c * 12 + x0c) * 2 + 0]
              + lx * (1.f - ly)         * r3[(y0c * 12 + x1c) * 2 + 0]
              + (1.f - lx) * ly         * r3[(y1c * 12 + x0c) * 2 + 0]
              + lx * ly                 * r3[(y1c * 12 + x1c) * 2 + 0];
    float p3y = (1.f - lx) * (1.f - ly) * r3[(y0c * 12 + x0c) * 2 + 1]
              + lx * (1.f - ly)         * r3[(y0c * 12 + x1c) * 2 + 1]
              + (1.f - lx) * ly         * r3[(y1c * 12 + x0c) * 2 + 1]
              + lx * ly                 * r3[(y1c * 12 + x1c) * 2 + 1];
    ref[((size_t)b * 576 + q) * 2 + 0] = (p1x + p2x + p3x) * (1.f / 3.f);
    ref[((size_t)b * 576 + q) * 2 + 1] = (p1y + p2y + p3y) * (1.f / 3.f);
  }
}

// ---------- WMMA bf16 GEMM: out[M x (NT*16)] = A[M x (KT*32)](bf16) * Wpack + bias ----------
// A layout (contiguous k, same as B pack) 16x32: lane(&15)=M row; hi=lane>>4 selects K+16;
// VGPR v: k-offset 2v + 16*hi.  C/D: VGPR r -> M=r+8*hi, N=lane&15.
// Fully unrolled K loop; all loads are immediate offsets from per-lane base pointers.
template <int NTPW, int STORE_BF16, int NT, int KT>
__global__ void gemm_wmma_kernel(const unsigned short* __restrict__ A,
                                 const unsigned int* __restrict__ Wp,
                                 const float* __restrict__ bias,
                                 void* __restrict__ out,
                                 int M) {
  const int Npad = NT * 16;
  const int K    = KT * 32;
  const int NG   = NT / NTPW;
  int wave  = blockIdx.x * (blockDim.x >> 5) + (threadIdx.x >> 5);
  int mtile = wave / NG;
  int ng    = wave % NG;
  if (mtile * 16 >= M) return;  // wave-uniform exit keeps EXEC all-ones for WMMA
  int lane = threadIdx.x & 31;
  int mr = lane & 15, hi = lane >> 4;

  union Frag { v16bf v; uint4 q[2]; };
  v8f zero = {};
  v8f acc[NTPW];
#pragma unroll
  for (int t = 0; t < NTPW; ++t) acc[t] = zero;

  // Per-lane A base: row (mtile*16+mr), hi half folded in (+16 elements = +32B).
  const uint4* abase = (const uint4*)(A + (size_t)(mtile * 16 + mr) * K + 16 * hi);
  // Per-lane B bases (one per N-tile); K-step stride = NT*1024 bytes = NT*64 uint4.
  const uint4* bbase[NTPW];
#pragma unroll
  for (int t = 0; t < NTPW; ++t)
    bbase[t] = (const uint4*)(Wp + ((size_t)(ng * NTPW + t) * 32 + lane) * 8);

#pragma unroll
  for (int kt = 0; kt < KT; ++kt) {
    Frag a;
    a.q[0] = abase[kt * 4 + 0];  // elements kt*32 + {0..7}(+16hi)
    a.q[1] = abase[kt * 4 + 1];  // elements kt*32 + 8 + {0..7}(+16hi)
#pragma unroll
    for (int t = 0; t < NTPW; ++t) {
      Frag b;
      b.q[0] = bbase[t][kt * NT * 64 + 0];
      b.q[1] = bbase[t][kt * NT * 64 + 1];
      acc[t] = __builtin_amdgcn_wmma_f32_16x16x32_bf16(
          false, a.v, false, b.v, (short)0, acc[t], false, false);
    }
  }

#pragma unroll
  for (int t = 0; t < NTPW; ++t) {
    int n = (ng * NTPW + t) * 16 + mr;
    float bv = bias ? bias[n] : 0.f;
#pragma unroll
    for (int r = 0; r < 8; ++r) {
      int row = mtile * 16 + r + 8 * hi;
      float o = acc[t][r] + bv;
      if (row < M) {
        if (STORE_BF16)
          ((unsigned short*)out)[(size_t)row * Npad + n] = f32_to_bf16(o);
        else
          ((float*)out)[(size_t)row * Npad + n] = o;
      }
    }
  }
}

// ---------- deformable sampling: 64 lanes per (b,q,head), lane = channel ----------
__global__ void sample_kernel(const unsigned short* __restrict__ val,
                              const float* __restrict__ qproj,
                              const float* __restrict__ ref,
                              unsigned short* __restrict__ outacc) {
  int unit = blockIdx.x * (blockDim.x >> 6) + (threadIdx.x >> 6);
  if (unit >= 32 * 576 * 6) return;
  int c  = threadIdx.x & 63;
  int h  = unit % 6;
  int bq = unit / 6;
  int b  = bq / 576;
  const float* qp = qproj + (size_t)bq * 224;
  float rx = ref[bq * 2 + 0];
  float ry = ref[bq * 2 + 1];
  // softmax over NL*NP = 12 logits for this head
  float logit[12];
  float mx = -3.0e38f;
#pragma unroll
  for (int i = 0; i < 12; ++i) { logit[i] = qp[144 + h * 12 + i]; mx = fmaxf(mx, logit[i]); }
  float s = 0.f;
#pragma unroll
  for (int i = 0; i < 12; ++i) { logit[i] = __expf(logit[i] - mx); s += logit[i]; }
  float inv = 1.f / s;
  const int starts[3] = {0, 2304, 2880};
  const int dims[3]   = {48, 24, 12};
  float acc = 0.f;
#pragma unroll
  for (int l = 0; l < 3; ++l) {
    int Wl = dims[l], Hl = dims[l];
    size_t lvl_base = (size_t)b * 3024 + starts[l];
#pragma unroll
    for (int p = 0; p < 4; ++p) {
      float ox = qp[h * 24 + l * 8 + p * 2 + 0];
      float oy = qp[h * 24 + l * 8 + p * 2 + 1];
      float aw = logit[l * 4 + p] * inv;
      float px = (ox + rx) * Wl - 0.5f;
      float py = (oy + ry) * Hl - 0.5f;
      float fx = floorf(px), fy = floorf(py);
      int x0 = (int)fx, y0 = (int)fy;
      float lx = px - fx, ly = py - fy;
      float w[4]  = {(1.f - lx) * (1.f - ly), lx * (1.f - ly), (1.f - lx) * ly, lx * ly};
      int   xs[4] = {x0, x0 + 1, x0, x0 + 1};
      int   ys[4] = {y0, y0, y0 + 1, y0 + 1};
      float sv = 0.f;
#pragma unroll
      for (int k = 0; k < 4; ++k) {
        int xi = xs[k], yi = ys[k];
        if (xi >= 0 && xi < Wl && yi >= 0 && yi < Hl) {
          size_t idx = ((lvl_base + (size_t)yi * Wl + xi) * 6 + h) * 64 + c;
          sv += w[k] * bf16_to_f32(val[idx]);
        }
      }
      acc += aw * sv;
    }
  }
  outacc[(size_t)unit * 64 + c] = f32_to_bf16(acc);
}

extern "C" void kernel_launch(void* const* d_in, const int* in_sizes, int n_in,
                              void* d_out, int out_size, void* d_ws, size_t ws_size,
                              hipStream_t stream) {
  const float* query  = (const float*)d_in[0];
  const float* value  = (const float*)d_in[1];
  const float* refpts = (const float*)d_in[2];
  const float* Wv     = (const float*)d_in[3];
  const float* bv     = (const float*)d_in[4];
  const float* Woff   = (const float*)d_in[5];
  const float* boff   = (const float*)d_in[6];
  const float* Wattn  = (const float*)d_in[7];
  const float* battn  = (const float*)d_in[8];
  const float* Wout   = (const float*)d_in[9];
  const float* bout   = (const float*)d_in[10];
  float* out = (float*)d_out;

  const int B = 32, Lq = 576, Lv = 3024, C = 384;
  const size_t nVal = (size_t)B * Lv * C;  // 37,158,912
  const size_t nQ   = (size_t)B * Lq * C;  // 7,077,888

  char* ws = (char*)d_ws;
  size_t cur = 0;
  auto alloc = [&](size_t bytes) -> char* {
    char* p = ws + cur;
    cur += (bytes + 255) & ~(size_t)255;
    return p;
  };
  unsigned short* valb    = (unsigned short*)alloc(nVal * 2);
  unsigned short* qb      = (unsigned short*)alloc(nQ * 2);
  unsigned short* valproj = (unsigned short*)alloc(nVal * 2);
  float*          qproj   = (float*)alloc((size_t)B * Lq * 224 * 4);
  float*          refw    = (float*)alloc((size_t)B * Lq * 2 * 4);
  unsigned short* outacc  = (unsigned short*)alloc(nQ * 2);
  unsigned int*   wvp     = (unsigned int*)alloc((size_t)12 * 24 * 32 * 8 * 4);
  float*          wcat    = (float*)alloc((size_t)384 * 224 * 4);
  unsigned int*   wcatp   = (unsigned int*)alloc((size_t)12 * 14 * 32 * 8 * 4);
  unsigned int*   woutp   = (unsigned int*)alloc((size_t)12 * 24 * 32 * 8 * 4);
  float*          bcat    = (float*)alloc(224 * 4);
  (void)ws_size; (void)in_sizes; (void)n_in; (void)out_size;

  cvt_bf16_kernel<<<4096, 256, 0, stream>>>(value, valb, nVal);
  cvt_bf16_kernel<<<2048, 256, 0, stream>>>(query, qb, nQ);
  build_wcat_kernel<<<(384 * 224 + 255) / 256, 256, 0, stream>>>(Woff, boff, Wattn, battn, wcat, bcat);
  pack_w_kernel<<<(12 * 24 * 32 * 8 + 255) / 256, 256, 0, stream>>>(Wv, wvp, 384, 384);
  pack_w_kernel<<<(12 * 14 * 32 * 8 + 255) / 256, 256, 0, stream>>>(wcat, wcatp, 384, 224);
  pack_w_kernel<<<(12 * 24 * 32 * 8 + 255) / 256, 256, 0, stream>>>(Wout, woutp, 384, 384);
  ref_kernel<<<B, 256, 0, stream>>>(refpts, refw);

  {  // val = value @ W_value + b_value  (M=96768, N=384), bf16 out
    int M = B * Lv;
    int NG = 24 / 4;
    int waves = (M / 16) * NG;
    int blocks = (waves * 32 + 255) / 256;
    gemm_wmma_kernel<4, 1, 24, 12><<<blocks, 256, 0, stream>>>(valb, wvp, bv, valproj, M);
  }
  {  // qproj = query @ [W_off|W_attn] (M=18432, Npad=224), f32 out
    int M = B * Lq;
    int NG = 14 / 2;
    int waves = (M / 16) * NG;
    int blocks = (waves * 32 + 255) / 256;
    gemm_wmma_kernel<2, 0, 14, 12><<<blocks, 256, 0, stream>>>(qb, wcatp, bcat, qproj, M);
  }
  {  // deformable bilinear sampling + head accumulation
    int units = B * Lq * 6;
    int blocks = (units + 3) / 4;
    sample_kernel<<<blocks, 256, 0, stream>>>(valproj, qproj, refw, outacc);
  }
  {  // out = out_acc @ W_out + b_out  (M=18432, N=384), f32 out to d_out
    int M = B * Lq;
    int NG = 24 / 4;
    int waves = (M / 16) * NG;
    int blocks = (waves * 32 + 255) / 256;
    gemm_wmma_kernel<4, 0, 24, 12><<<blocks, 256, 0, stream>>>(outacc, woutp, bout, out, M);
  }
}
